// Head_34823594836117
// MI455X (gfx1250) — matrix operand order repaired and checked by reading, not compiled, not run on hardware
//
#include <hip/hip_runtime.h>

typedef __attribute__((ext_vector_type(16))) _Float16 v16h;
typedef __attribute__((ext_vector_type(8)))  _Float16 v8h;
typedef __attribute__((ext_vector_type(4)))  _Float16 v4h;
typedef __attribute__((ext_vector_type(8)))  float    v8f;
typedef __attribute__((ext_vector_type(4)))  float    v4f;

typedef __attribute__((address_space(3))) _Float16 lds_f16;

#define BATCH 8
#define TSEQ  2048
#define CEMB  1024
#define HDIM  64

__device__ __forceinline__ v8f wmma_f16(v16h a, v16h b, v8f c) {
  return __builtin_amdgcn_wmma_f32_16x16x32_f16(false, a, false, b, (short)0, c,
                                                false, false);
}

// A-fragment (16x32 f16): 16 contiguous K values per lane, K base+2*hi8 .. +15
__device__ __forceinline__ v16h ldA(const _Float16* p, int hi8) {
  v8h lo = *(const v8h*)(p + 2 * hi8);
  v8h h2 = *(const v8h*)(p + 2 * hi8 + 8);
  v16h r;
#pragma unroll
  for (int i = 0; i < 8; ++i) { r[i] = lo[i]; r[8 + i] = h2[i]; }
  return r;
}

// B-fragment (32x16 f16): 16 contiguous K values per lane (caller adds hi?16:0)
__device__ __forceinline__ v16h ldB(const _Float16* p) {
  v8h lo = *(const v8h*)(p);
  v8h h2 = *(const v8h*)(p + 8);
  v16h r;
#pragma unroll
  for (int i = 0; i < 8; ++i) { r[i] = lo[i]; r[8 + i] = h2[i]; }
  return r;
}

// Async global -> LDS staging of one 16-byte chunk per lane (gfx1250 path)
__device__ __forceinline__ void async_b128(const _Float16* gptr, _Float16* lptr) {
  lds_f16* l = (lds_f16*)lptr;
  asm volatile("global_load_async_to_lds_b128 %0, %1, off"
               :: "v"(l), "v"(gptr) : "memory");
}
__device__ __forceinline__ void async_wait0() {
  asm volatile("s_wait_asynccnt 0x0" ::: "memory");
}

// ---------------- Kernel 0: W [1024,64] f32 -> Wt [64,1024] f16 ----------------
__global__ __launch_bounds__(256) void w_transpose_f16(const float* __restrict__ W,
                                                       _Float16* __restrict__ Wt) {
  int idx = blockIdx.x * 256 + threadIdx.x;   // 0..65535
  int n = idx >> 10;
  int k = idx & 1023;
  Wt[idx] = (_Float16)W[k * HDIM + n];
}

// ------------- Kernel 1: fused Q/K/V projection -------------
// grid = 1024 blocks (16 tokens each), block = 128 (4 waves; wave w owns n-tile w).
// x slice is double-buffered in LDS; the next slice's global load is issued
// before the barrier so it overlaps the current step's 3 WMMAs.
__global__ __launch_bounds__(128) void proj_qkv(const float* __restrict__ x,
                                                const _Float16* __restrict__ Wt,
                                                _Float16* __restrict__ qh,
                                                _Float16* __restrict__ kh,
                                                _Float16* __restrict__ vT) {
  __shared__ _Float16 xt[2][16][32];   // 2 KB: double-buffered 16x32 x slice (f16)

  const int tid  = threadIdx.x;
  const int lane = tid & 31;
  const int wv   = tid >> 5;        // 0..3 -> n-tile of 16
  const int ln   = lane & 15;
  const int hi   = lane >> 4;
  const int hi8  = hi * 8;
  const int ho   = hi ? 16 : 0;
  const int m0   = blockIdx.x * 16;

  // cooperative staging coords: 128 threads x 4 floats = 16x32 slice
  const int srow = tid >> 3;        // 0..15
  const int scol = (tid & 7) * 4;   // 0..28
  const float* xs = x + (size_t)(m0 + srow) * CEMB + scol;

  const _Float16* wq  = Wt + ((size_t)(0 * HDIM + wv * 16 + ln)) * CEMB;
  const _Float16* wk  = Wt + ((size_t)(1 * HDIM + wv * 16 + ln)) * CEMB;
  const _Float16* wvp = Wt + ((size_t)(2 * HDIM + wv * 16 + ln)) * CEMB;

  v8f aq = {}, ak = {}, av = {};
  v4f c = *(const v4f*)(xs);          // prefetch first slice into registers
  for (int kk = 0; kk < CEMB; kk += 32) {
    const int ib = (kk >> 5) & 1;
    v4h h;
#pragma unroll
    for (int i = 0; i < 4; ++i) h[i] = (_Float16)c[i];
    *(v4h*)&xt[ib][srow][scol] = h;
    if (kk + 32 < CEMB) c = *(const v4f*)(xs + kk + 32);   // prefetch next slice
    __syncthreads();

    v16h a = ldA(&xt[ib][ln][0], hi8);
    aq = wmma_f16(a, ldB(wq  + kk + ho), aq);
    ak = wmma_f16(a, ldB(wk  + kk + ho), ak);
    av = wmma_f16(a, ldB(wvp + kk + ho), av);
    // no trailing barrier: next iteration writes the OTHER buffer; the
    // following top barrier orders reuse of this one.
  }

  const int col = wv * 16 + ln;
#pragma unroll
  for (int r = 0; r < 8; ++r) {
    int m = m0 + r + hi8;
    qh[(size_t)m * HDIM + col] = (_Float16)aq[r];
    kh[(size_t)m * HDIM + col] = (_Float16)ak[r];
    int b = m >> 11;
    int t = m & (TSEQ - 1);
    vT[((size_t)(b * HDIM + col)) * TSEQ + t] = (_Float16)av[r];
  }
}

// ------------- Kernel 2: flash attention -------------
// grid = 128 blocks, block = 256 (8 waves; each wave owns a 16-query tile).
// K/V tiles are double-buffered in LDS; async global->LDS loads for tile i+1
// are in flight during all of tile i's compute (ASYNCcnt pipelining).
__global__ __launch_bounds__(256) void attn(const _Float16* __restrict__ qh,
                                            const _Float16* __restrict__ kh,
                                            const _Float16* __restrict__ vT,
                                            float* __restrict__ out) {
  __shared__ _Float16 kt[2][32][64];    // 8 KB: 32 keys x 64 head dims, x2
  __shared__ _Float16 vt[2][64][32];    // 8 KB: 64 head dims x 32 keys, x2
  __shared__ _Float16 ldsP[8][16][32];  // 8 KB: per-wave P transpose staging

  const int tid  = threadIdx.x;
  const int lane = tid & 31;
  const int wv   = tid >> 5;    // 0..7
  const int ln   = lane & 15;
  const int hi   = lane >> 4;
  const int hi8  = hi * 8;
  const int ho   = hi ? 16 : 0;
  const int m0   = (blockIdx.x * 8 + wv) * 16;  // global query tile base
  const int b    = m0 >> 11;

  const _Float16* kbase = kh + (size_t)b * TSEQ * HDIM;
  const _Float16* vbase = vT + (size_t)b * HDIM * TSEQ;

  // staging coords: one async B128 per thread per tile
  const int krow = tid >> 3;         // 0..31
  const int kcol = (tid & 7) * 8;    // 0..56
  const int vrow = tid >> 2;         // 0..63
  const int vcol = (tid & 3) * 8;    // 0..24
  const _Float16* kg = kbase + (size_t)krow * HDIM + kcol;
  const _Float16* vg = vbase + (size_t)vrow * TSEQ + vcol;

  // Q fragments (two K-steps over head dim 64)
  const _Float16* qp = qh + (size_t)(m0 + ln) * HDIM;
  v16h aq0 = ldA(qp, hi8);
  v16h aq1 = ldA(qp + 32, hi8);

  float mrow[8], lrow[8];
#pragma unroll
  for (int r = 0; r < 8; ++r) { mrow[r] = -1e30f; lrow[r] = 0.0f; }
  v8f o0 = {}, o1 = {}, o2 = {}, o3 = {};

  const float scale = 0.03125f;   // 1024^-0.5

  // prologue: kick off tile 0
  async_b128(kg, &kt[0][krow][kcol]);
  async_b128(vg, &vt[0][vrow][vcol]);

  for (int kb = 0; kb < TSEQ; kb += 32) {
    const int ib = (kb >> 5) & 1;

    // wait for the prefetch of THIS tile, then sync the block
    async_wait0();
    __syncthreads();

    // immediately launch async prefetch of the NEXT tile (alternate buffer)
    if (kb + 32 < TSEQ) {
      async_b128(kg + (size_t)(kb + 32) * HDIM, &kt[ib ^ 1][krow][kcol]);
      async_b128(vg + (kb + 32),                &vt[ib ^ 1][vrow][vcol]);
    }

    // ---- S = Q @ K^T for 32 keys (from LDS) ----
    v8f s0 = {}, s1 = {};
    {
      const _Float16* kp0 = &kt[ib][ln][0] + ho;
      const _Float16* kp1 = &kt[ib][16 + ln][0] + ho;
      s0 = wmma_f16(aq0, ldB(kp0), s0);
      s0 = wmma_f16(aq1, ldB(kp0 + 32), s0);
      s1 = wmma_f16(aq0, ldB(kp1), s1);
      s1 = wmma_f16(aq1, ldB(kp1 + 32), s1);
    }

    // ---- online softmax; write P (f16) into per-wave LDS slab ----
#pragma unroll
    for (int r = 0; r < 8; ++r) {
      float a0 = s0[r] * scale;
      float a1 = s1[r] * scale;
      float mx = fmaxf(a0, a1);
      mx = fmaxf(mx, __shfl_xor(mx, 1, 32));
      mx = fmaxf(mx, __shfl_xor(mx, 2, 32));
      mx = fmaxf(mx, __shfl_xor(mx, 4, 32));
      mx = fmaxf(mx, __shfl_xor(mx, 8, 32));
      float mn = fmaxf(mrow[r], mx);
      float alpha = __expf(mrow[r] - mn);
      float p0 = __expf(a0 - mn);
      float p1 = __expf(a1 - mn);
      float sum = p0 + p1;
      sum += __shfl_xor(sum, 1, 32);
      sum += __shfl_xor(sum, 2, 32);
      sum += __shfl_xor(sum, 4, 32);
      sum += __shfl_xor(sum, 8, 32);
      lrow[r] = lrow[r] * alpha + sum;
      mrow[r] = mn;
      o0[r] *= alpha; o1[r] *= alpha; o2[r] *= alpha; o3[r] *= alpha;
      ldsP[wv][r + hi8][ln]      = (_Float16)p0;
      ldsP[wv][r + hi8][16 + ln] = (_Float16)p1;
    }
    __syncthreads();

    // ---- O += P @ V (P from LDS in A layout, V from staged LDS tile) ----
    v16h ap = ldA(&ldsP[wv][ln][0], hi8);
    o0 = wmma_f16(ap, ldB(&vt[ib][0  + ln][0] + ho), o0);
    o1 = wmma_f16(ap, ldB(&vt[ib][16 + ln][0] + ho), o1);
    o2 = wmma_f16(ap, ldB(&vt[ib][32 + ln][0] + ho), o2);
    o3 = wmma_f16(ap, ldB(&vt[ib][48 + ln][0] + ho), o3);
    // no trailing barrier: next iteration's async writes hit the OTHER buffer,
    // and its top barrier orders reuse of this one (reads here complete before
    // this wave's next barrier signal via the wmma register dependency).
  }

  // ---- normalize and store fp32 output ----
#pragma unroll
  for (int r = 0; r < 8; ++r) {
    float inv = 1.0f / lrow[r];
    float* op = out + (size_t)(m0 + r + hi8) * HDIM + ln;
    op[0]  = o0[r] * inv;
    op[16] = o1[r] * inv;
    op[32] = o2[r] * inv;
    op[48] = o3[r] * inv;
  }
}

extern "C" void kernel_launch(void* const* d_in, const int* in_sizes, int n_in,
                              void* d_out, int out_size, void* d_ws, size_t ws_size,
                              hipStream_t stream) {
  (void)in_sizes; (void)n_in; (void)out_size; (void)ws_size;
  const float* x  = (const float*)d_in[0];
  const float* Wk = (const float*)d_in[1];
  const float* Wq = (const float*)d_in[2];
  const float* Wv = (const float*)d_in[3];

  char* ws = (char*)d_ws;
  _Float16* Wt  = (_Float16*)ws;                               // [3][64][1024] q,k,v
  _Float16* qhb = (_Float16*)(ws + 393216);                    // [16384][64]
  _Float16* khb = (_Float16*)(ws + 393216 + 2097152);          // [16384][64]
  _Float16* vTb = (_Float16*)(ws + 393216 + 2 * 2097152);      // [8][64][2048]
  float* outp = (float*)d_out;

  w_transpose_f16<<<256, 256, 0, stream>>>(Wq, Wt);
  w_transpose_f16<<<256, 256, 0, stream>>>(Wk, Wt + 65536);
  w_transpose_f16<<<256, 256, 0, stream>>>(Wv, Wt + 131072);
  proj_qkv<<<1024, 128, 0, stream>>>(x, Wt, qhb, khb, vTb);
  attn<<<128, 256, 0, stream>>>(qhb, khb, vTb, outp);
}
